// BondCenteredTensorMomentDescriptor_32435593019866
// MI455X (gfx1250) — matrix operand order repaired and checked by reading, not compiled, not run on hardware
//
#include <hip/hip_runtime.h>

typedef _Float16 v16h __attribute__((ext_vector_type(16)));
typedef _Float16 v8h  __attribute__((ext_vector_type(8)));
typedef float    v8f  __attribute__((ext_vector_type(8)));

#define NFEAT   16
#define NSPH    25
#define NPATH   65
#define KCHUNK  25      // K chunks of 32 (kk = a*32 + b, a,b in 0..24)
#define PAIRS_PER_BLK 32

// ---------------------------------------------------------------------------
// zero the workspace (B matrix fragments)
// ---------------------------------------------------------------------------
__global__ void zero_ws(unsigned int* w, int n) {
    int i = blockIdx.x * blockDim.x + threadIdx.x;
    if (i < n) w[i] = 0u;
}

// ---------------------------------------------------------------------------
// On-device Clebsch-Gordan (real spherical harmonics) -> pre-swizzled WMMA
// B fragments, folded with tp_weights per feature.
// ---------------------------------------------------------------------------
__device__ __forceinline__ int imax2(int a, int b) { return a > b ? a : b; }
__device__ __forceinline__ int imin2(int a, int b) { return a < b ? a : b; }
__device__ __forceinline__ int iabs2(int a)        { return a < 0 ? -a : a; }

__device__ double dfac(int n) {
    double r = 1.0;
    for (int k = 2; k <= n; ++k) r *= (double)k;
    return r;
}

__device__ double cg_complex(int j1, int m1, int j2, int m2, int j3, int m3) {
    if (m1 + m2 != m3) return 0.0;
    if (j3 < iabs2(j1 - j2) || j3 > j1 + j2) return 0.0;
    double pref = sqrt((double)(2 * j3 + 1) * dfac(j3 + j1 - j2) * dfac(j3 - j1 + j2) *
                       dfac(j1 + j2 - j3) / dfac(j1 + j2 + j3 + 1));
    pref *= sqrt(dfac(j3 + m3) * dfac(j3 - m3) * dfac(j1 - m1) * dfac(j1 + m1) *
                 dfac(j2 - m2) * dfac(j2 + m2));
    int kmin = imax2(0, imax2(j2 - j3 - m1, j1 + m2 - j3));
    int kmax = imin2(j1 + j2 - j3, imin2(j1 - m1, j2 + m2));
    double s = 0.0;
    for (int k = kmin; k <= kmax; ++k) {
        double den = dfac(k) * dfac(j1 + j2 - j3 - k) * dfac(j1 - m1 - k) *
                     dfac(j2 + m2 - k) * dfac(j3 - j2 + m1 + k) * dfac(j3 - j1 - m2 + k);
        s += ((k & 1) ? -1.0 : 1.0) / den;
    }
    return pref * s;
}

// Row i of the complex->real transform U(l): at most 2 nonzero columns.
__device__ int urow(int l, int i, int m[2], double re[2], double im[2]) {
    const double s2 = 0.70710678118654752440;
    int mi = i - l;
    if (mi == 0) { m[0] = 0; re[0] = 1.0; im[0] = 0.0; return 1; }
    if (mi > 0) {
        int mm = mi; double sg = (mm & 1) ? -1.0 : 1.0;
        m[0] =  mm; re[0] = sg * s2; im[0] = 0.0;   // U[l+m, l+m]
        m[1] = -mm; re[1] = s2;      im[1] = 0.0;   // U[l+m, l-m]
        return 2;
    }
    int mm = -mi; double sg = (mm & 1) ? -1.0 : 1.0;
    m[0] = -mm; re[0] = 0.0; im[0] = s2;            // U[l-m, l-m] =  i*s2
    m[1] =  mm; re[1] = 0.0; im[1] = -sg * s2;      // U[l-m, l+m] = -i*(-1)^m*s2
    return 2;
}

// one block per path; writes f16 B fragments: Bpk[((f*25+c)*4+nt)*32+lane][t]
__global__ __launch_bounds__(256)
void build_B_kernel(const float* __restrict__ tpw, _Float16* __restrict__ Bpk) {
    int pidx = blockIdx.x;
    int l1 = 0, l2 = 0, l3 = 0;
    {
        int cnt = 0; bool done = false;
        for (int a = 0; a <= 4 && !done; ++a)
            for (int b = 0; b <= 4 && !done; ++b) {
                int hi = imin2(4, a + b);
                for (int c = iabs2(a - b); c <= hi; ++c) {
                    if (cnt == pidx) { l1 = a; l2 = b; l3 = c; done = true; break; }
                    ++cnt;
                }
            }
    }
    const int n1 = 2 * l1 + 1, n2 = 2 * l2 + 1, n3 = 2 * l3 + 1;
    const int total = n1 * n2 * n3;

    __shared__ double sRe[729];
    __shared__ double sIm[729];
    __shared__ int pickReal;

    for (int e = threadIdx.x; e < total; e += blockDim.x) {
        int i = e / (n2 * n3);
        int j = (e / n3) % n2;
        int k = e % n3;
        int m1v[2], m2v[2], m3v[2];
        double r1[2], i1[2], r2[2], i2[2], r3[2], i3[2];
        int c1 = urow(l1, i, m1v, r1, i1);
        int c2 = urow(l2, j, m2v, r2, i2);
        int c3 = urow(l3, k, m3v, r3, i3);
        double accR = 0.0, accI = 0.0;
        for (int a = 0; a < c1; ++a)
            for (int b = 0; b < c2; ++b) {
                int m3 = m1v[a] + m2v[b];
                if (m3 < -l3 || m3 > l3) continue;
                for (int cc = 0; cc < c3; ++cc) {
                    if (m3v[cc] != m3) continue;
                    double cg = cg_complex(l1, m1v[a], l2, m2v[b], l3, m3);
                    if (cg == 0.0) continue;
                    double ur = r1[a] * r2[b] - i1[a] * i2[b];
                    double ui = r1[a] * i2[b] + i1[a] * r2[b];
                    double vr = ur * r3[cc] + ui * i3[cc];   // * conj(U3)
                    double vi = ui * r3[cc] - ur * i3[cc];
                    accR += vr * cg;
                    accI += vi * cg;
                }
            }
        sRe[e] = accR; sIm[e] = accI;
    }
    __syncthreads();
    if (threadIdx.x == 0) {
        double mr = 0.0, mi = 0.0;
        for (int e = 0; e < total; ++e) {
            mr = fmax(mr, fabs(sRe[e]));
            mi = fmax(mi, fabs(sIm[e]));
        }
        pickReal = (mr >= mi) ? 1 : 0;
    }
    __syncthreads();

    const int parity = (l1 + l2) & 1;
    for (int e = threadIdx.x; e < total; e += blockDim.x) {
        double val = pickReal ? sRe[e] : sIm[e];
        if (val == 0.0) continue;                 // Bpk pre-zeroed
        int i = e / (n2 * n3);
        int j = (e / n3) % n2;
        int k = e % n3;
        int aG = l1 * l1 + i;                     // chunk index (K high bits)
        int bG = l2 * l2 + j;                     // K within chunk
        int pc = parity * 25 + l3 * l3 + k;       // output channel
        int h  = bG >> 4, t = bG & 15;            // B layout: lanes 0-15 K=0..15, 16-31 K=16..31
        int nt = pc >> 4;
        int lane = h * 16 + (pc & 15);
        for (int f = 0; f < NFEAT; ++f) {
            float w = (float)val * tpw[pidx * NFEAT + f];
            long idx = ((((long)f * KCHUNK + aG) * 4 + nt) * 32 + lane) * 16 + t;
            Bpk[idx] = (_Float16)w;
        }
    }
}

// ---------------------------------------------------------------------------
// Main pair kernel: 32 pairs per block (2 M-tiles), 8 waves x 2 features.
// Descriptors staged coalesced -> LDS (f16); B fragments double-buffered.
// Per feature: 25 K-chunks x 4 N-tiles x 2 M-tiles WMMA.
// ---------------------------------------------------------------------------
__global__ __launch_bounds__(256)
void pair_kernel(const float* __restrict__ desc,
                 const float* __restrict__ disp,
                 const int*   __restrict__ nidx,
                 const _Float16* __restrict__ Bpk,
                 float* __restrict__ out) {
    // x2 rows padded to 32 halfs (64B) for aligned ds_load_b128 A-fragments.
    __shared__ _Float16 sX2[PAIRS_PER_BLK][NFEAT][32];   // 32 KB
    __shared__ _Float16 sX1[PAIRS_PER_BLK][NFEAT][NSPH]; // 25.6 KB
    __shared__ float    sY[PAIRS_PER_BLK][NSPH];         // Y(u)*cutoff, 3.2 KB
    __shared__ float    sG[PAIRS_PER_BLK][NFEAT];        // gaussians, 2 KB

    const int tid = threadIdx.x;
    const long pbase = (long)blockIdx.x * PAIRS_PER_BLK;

    // ---- per-pair radial & angular basis ----
    if (tid < PAIRS_PER_BLK) {
        long p = pbase + tid;
        float dx = disp[3 * p + 0], dy = disp[3 * p + 1], dz = disp[3 * p + 2];
        float r = sqrtf(dx * dx + dy * dy + dz * dz);
        float inv = 1.0f / (r + 1e-12f);
        float z = dz * inv;
        float sin_t = sqrtf(fmaxf(1.0f - z * z, 0.0f));
        float phi = atan2f(dy * inv, dx * inv);
        float P[5][5];
        P[0][0] = 1.0f;
        for (int m = 1; m <= 4; ++m) P[m][m] = -(2.0f * m - 1.0f) * sin_t * P[m - 1][m - 1];
        for (int m = 0; m < 4; ++m)  P[m + 1][m] = (2.0f * m + 1.0f) * z * P[m][m];
        for (int m = 0; m <= 4; ++m)
            for (int l = m + 2; l <= 4; ++l)
                P[l][m] = ((2.0f * l - 1.0f) * z * P[l - 1][m] - (float)(l + m - 1) * P[l - 2][m]) /
                          (float)(l - m);
        float cut = (r < 5.0f) ? 0.5f * (cosf(3.14159265358979f * r / 5.0f) + 1.0f) : 0.0f;
        const float FOURPI = 12.566370614359172f;
        const float fct[9] = {1.f, 1.f, 2.f, 6.f, 24.f, 120.f, 720.f, 5040.f, 40320.f};
        const float SQ2 = 1.41421356237309505f;
        int s = 0;
        for (int l = 0; l <= 4; ++l) {
            for (int mm = -l; mm <= l; ++mm) {
                int am = mm < 0 ? -mm : mm;
                float N = sqrtf((2.0f * l + 1.0f) / FOURPI * fct[l - am] / fct[l + am]);
                float v;
                if (mm < 0)       v = SQ2 * N * P[l][am] * sinf((float)am * phi);
                else if (mm == 0) v = N * P[l][0];
                else              v = SQ2 * N * P[l][mm] * cosf((float)mm * phi);
                sY[tid][s] = v * cut;
                ++s;
            }
        }
        for (int fi = 0; fi < NFEAT; ++fi) {
            float mu = 5.0f * (float)fi / 15.0f;   // linspace(0, CUTOFF, 16)
            float d = 3.2f * (r - mu);             // gamma = F/CUTOFF
            sG[tid][fi] = expf(-d * d);
        }
    }

    // ---- zero the x2 pad region (sph 25..31) so no NaN enters A fragments ----
    for (int idx = tid; idx < PAIRS_PER_BLK * NFEAT * 7; idx += 256) {
        int pr = idx / (NFEAT * 7);
        int rm = idx % (NFEAT * 7);
        sX2[pr][rm / 7][25 + rm % 7] = (_Float16)0.0f;
    }

    // ---- stage descriptors: coalesced float4 global loads -> f16 LDS ----
    // 32 pairs x 25 sph x 16 feat floats per side; float4 along feature dim.
    for (int idx = tid; idx < PAIRS_PER_BLK * NSPH * 4; idx += 256) {
        int pr = idx / (NSPH * 4);
        int rm = idx % (NSPH * 4);
        int s  = rm >> 2;
        int fg = (rm & 3) * 4;
        long p = pbase + pr;
        int a1 = nidx[2 * p + 0];
        int a2 = nidx[2 * p + 1];
        float4 v1 = *(const float4*)(desc + (long)a1 * (NSPH * NFEAT) + s * NFEAT + fg);
        float4 v2 = *(const float4*)(desc + (long)a2 * (NSPH * NFEAT) + s * NFEAT + fg);
        sX1[pr][fg + 0][s] = (_Float16)v1.x;
        sX1[pr][fg + 1][s] = (_Float16)v1.y;
        sX1[pr][fg + 2][s] = (_Float16)v1.z;
        sX1[pr][fg + 3][s] = (_Float16)v1.w;
        sX2[pr][fg + 0][s] = (_Float16)v2.x;
        sX2[pr][fg + 1][s] = (_Float16)v2.y;
        sX2[pr][fg + 2][s] = (_Float16)v2.z;
        sX2[pr][fg + 3][s] = (_Float16)v2.w;
    }
    __syncthreads();

    const int wave = tid >> 5;
    const int lane = tid & 31;
    const int row  = lane & 15;    // A-fragment M row; also C/D column N
    const int h    = lane >> 4;
    const int p0   = row;          // M-tile 0 pair (in-block)
    const int p1   = row + 16;     // M-tile 1 pair (in-block)
    const v16h* bbase = (const v16h*)Bpk;

    for (int fi = 0; fi < 2; ++fi) {
        const int f = wave * 2 + fi;

        // A-fragment x2 pattern: jj=16h+t (contig)
        v8h lo0 = *(const v8h*)&sX2[p0][f][16 * h];
        v8h hi0 = *(const v8h*)&sX2[p0][f][16 * h + 8];
        v8h lo1 = *(const v8h*)&sX2[p1][f][16 * h];
        v8h hi1 = *(const v8h*)&sX2[p1][f][16 * h + 8];
        v16h x2f0 = __builtin_shufflevector(lo0, hi0, 0,1,2,3,4,5,6,7,8,9,10,11,12,13,14,15);
        v16h x2f1 = __builtin_shufflevector(lo1, hi1, 0,1,2,3,4,5,6,7,8,9,10,11,12,13,14,15);

        v8f acc[8] = {};   // [mtile][ntile] -> acc[mtile*4+nt]

        // double-buffered B fragments
        long base = ((long)f * KCHUNK) * 4 * 32 + lane;
        v16h b0 = bbase[base];
        v16h b1 = bbase[base + 32];
        v16h b2 = bbase[base + 64];
        v16h b3 = bbase[base + 96];
        _Float16 x10 = sX1[p0][f][0];
        _Float16 x11 = sX1[p1][f][0];

        for (int c = 0; c < KCHUNK; ++c) {
            v16h n0, n1, n2, n3;
            _Float16 y0, y1;
            if (c < KCHUNK - 1) {
                long nb = ((long)f * KCHUNK + c + 1) * 4 * 32 + lane;
                n0 = bbase[nb];
                n1 = bbase[nb + 32];
                n2 = bbase[nb + 64];
                n3 = bbase[nb + 96];
                y0 = sX1[p0][f][c + 1];
                y1 = sX1[p1][f][c + 1];
            }
            v16h a0 = x2f0 * x10;   // W[row, c*32+jj] outer product, tile 0
            v16h a1 = x2f1 * x11;   // tile 1
            acc[0] = __builtin_amdgcn_wmma_f32_16x16x32_f16(false, a0, false, b0, (short)0, acc[0], false, false);
            acc[4] = __builtin_amdgcn_wmma_f32_16x16x32_f16(false, a1, false, b0, (short)0, acc[4], false, false);
            acc[1] = __builtin_amdgcn_wmma_f32_16x16x32_f16(false, a0, false, b1, (short)0, acc[1], false, false);
            acc[5] = __builtin_amdgcn_wmma_f32_16x16x32_f16(false, a1, false, b1, (short)0, acc[5], false, false);
            acc[2] = __builtin_amdgcn_wmma_f32_16x16x32_f16(false, a0, false, b2, (short)0, acc[2], false, false);
            acc[6] = __builtin_amdgcn_wmma_f32_16x16x32_f16(false, a1, false, b2, (short)0, acc[6], false, false);
            acc[3] = __builtin_amdgcn_wmma_f32_16x16x32_f16(false, a0, false, b3, (short)0, acc[3], false, false);
            acc[7] = __builtin_amdgcn_wmma_f32_16x16x32_f16(false, a1, false, b3, (short)0, acc[7], false, false);
            if (c < KCHUNK - 1) {
                b0 = n0; b1 = n1; b2 = n2; b3 = n3;
                x10 = y0; x11 = y1;
            }
        }

        // ---- epilogue: C/D layout (lane -> N=lane&15, VGPR g -> M=g+8h) ----
#pragma unroll
        for (int mt = 0; mt < 2; ++mt) {
#pragma unroll
            for (int nt = 0; nt < 4; ++nt) {
                int pc = nt * 16 + row;      // row = lane&15 acts as N here
                if (pc >= 50) continue;
                int parity = (pc >= 25) ? 1 : 0;
                int s = pc - 25 * parity;
                // parity of l(s): l=0:0 l=1:1 l=2:0 l=3:1 l=4:0
                int lpar = (s == 0) ? 0 : (s < 4) ? 1 : (s < 9) ? 0 : (s < 16) ? 1 : 0;
#pragma unroll
                for (int g = 0; g < 8; ++g) {
                    int m = mt * 16 + g + h * 8;
                    long pout = pbase + m;
                    float v = acc[mt * 4 + nt][g];
                    if (lpar == parity) v += sY[m][s] * sG[m][f];
                    out[((pout * 2 + parity) * NSPH + s) * NFEAT + f] = v;
                }
            }
        }
    }
}

// ---------------------------------------------------------------------------
extern "C" void kernel_launch(void* const* d_in, const int* in_sizes, int n_in,
                              void* d_out, int out_size, void* d_ws, size_t ws_size,
                              hipStream_t stream) {
    const float* desc = (const float*)d_in[0];   // (NUM_ATOMS, 1, 25, 16) f32
    const float* disp = (const float*)d_in[1];   // (NUM_PAIRS, 3) f32
    const float* tpw  = (const float*)d_in[2];   // (NPATH, 16) f32
    const int*   nidx = (const int*)d_in[3];     // (NUM_PAIRS, 2) i32
    float* out = (float*)d_out;

    _Float16* Bpk = (_Float16*)d_ws;             // 16*25*4*32*16 halfs = 1.6 MB
    const int nHalf = NFEAT * KCHUNK * 4 * 32 * 16;
    const int nWords = nHalf / 2;

    zero_ws<<<(nWords + 255) / 256, 256, 0, stream>>>((unsigned int*)d_ws, nWords);
    build_B_kernel<<<NPATH, 256, 0, stream>>>(tpw, Bpk);

    const int npairs = in_sizes[3] / 2;          // 100000
    pair_kernel<<<npairs / PAIRS_PER_BLK, 256, 0, stream>>>(desc, disp, nidx, Bpk, out);
}
